// DependencyParser_75703093559392
// MI455X (gfx1250) — matrix operand order repaired and checked by reading, not compiled, not run on hardware
//
#include <hip/hip_runtime.h>
#include <hip/hip_bf16.h>
#include <math.h>

// ---------------------------------------------------------------------------
// DependencyParser: 4x bi-LSTM (T=512, H=512) + all-pairs tanh edge scorer.
//   * per-layer input projections X @ Wih^T + b  -> bf16 WMMA GEMM (f32 acc)
//   * sequential LSTM recurrence                 -> fp32 VALU, h in LDS
//   * heads/mods projections                     -> bf16 WMMA GEMM
//   * pair scorer w.tanh(h_i + m_j)              -> VALU/TRANS, LDS tiles
// Ragged-K operands (K=449) are zero-padded to K=480 so the GEMM inner loop
// is branch-free; each wave computes a 16x32 C tile = 2 WMMAs per A fragment.
// ---------------------------------------------------------------------------

#define T_LEN 512
#define HDIM  512
#define G4H   2048     // 4*HDIM
#define H2    256
#define NP    513      // T_LEN + 1 (ROOT row)
#define KP0   480      // 449 padded up to a multiple of 32

typedef __attribute__((ext_vector_type(16))) __bf16 v16bf;
typedef __attribute__((ext_vector_type(8)))  float  v8f;

__device__ __forceinline__ v16bf pack_bf16(const float4 x, const float4 y,
                                           const float4 z, const float4 w)
{
    v16bf f;
    f[0]=(__bf16)x.x;  f[1]=(__bf16)x.y;  f[2]=(__bf16)x.z;  f[3]=(__bf16)x.w;
    f[4]=(__bf16)y.x;  f[5]=(__bf16)y.y;  f[6]=(__bf16)y.z;  f[7]=(__bf16)y.w;
    f[8]=(__bf16)z.x;  f[9]=(__bf16)z.y;  f[10]=(__bf16)z.z; f[11]=(__bf16)z.w;
    f[12]=(__bf16)w.x; f[13]=(__bf16)w.y; f[14]=(__bf16)w.z; f[15]=(__bf16)w.w;
    return f;
}

// ---------------------------------------------------------------------------
// Guard-free GEMM:  C[M,N] = A[M,K] @ W[N,K]^T + bias[N]
// Preconditions: K % 32 == 0, lda/ldw % 4 == 0 (16B-aligned rows), N % 16 == 0.
// One wave per 16x32 C tile: one A fragment feeds two v_wmma_f32_16x16x32_bf16
// (independent accumulator chains -> no WMMA->WMMA RAW hazard).
// Out-of-range M/N lanes are clamped; only the store is guarded.
// ---------------------------------------------------------------------------
__global__ __launch_bounds__(128)
void k_gemm_bias(const float* __restrict__ A, int lda,
                 const float* __restrict__ W, int ldw,
                 const float* __restrict__ bias,
                 float* __restrict__ C, int ldc,
                 int M, int N, int K)
{
    const int lane   = threadIdx.x & 31;
    const int wave   = threadIdx.x >> 5;
    const int tilesN = (N + 31) >> 5;             // 32-wide N tiles
    const int tilesM = (M + 15) >> 4;
    const int tile   = blockIdx.x * 4 + wave;
    if (tile >= tilesM * tilesN) return;          // whole-wave uniform exit

    const int tm  = tile / tilesN;
    const int tn  = tile % tilesN;
    const int g   = lane >> 4;                    // lane group 0/1
    const int l16 = lane & 15;

    int mrow  = tm * 16 + l16;       if (mrow  > M - 1) mrow  = M - 1;  // clamp
    int ncol0 = tn * 32 + l16;       if (ncol0 > N - 1) ncol0 = N - 1;
    int ncol1 = tn * 32 + 16 + l16;  if (ncol1 > N - 1) ncol1 = N - 1;
    const float* __restrict__ Arow  = A + (size_t)mrow  * lda;
    const float* __restrict__ Wrow0 = W + (size_t)ncol0 * ldw;
    const float* __restrict__ Wrow1 = W + (size_t)ncol1 * ldw;

    v8f acc0 = {}, acc1 = {};
    for (int k0 = 0; k0 < K; k0 += 32) {
        // prefetch next K block (speculative; dropped if past end of buffer)
        __builtin_prefetch(Arow  + k0 + 32, 0, 1);
        __builtin_prefetch(Wrow0 + k0 + 32, 0, 1);
        __builtin_prefetch(Wrow1 + k0 + 32, 0, 1);

        // A fragment (16x32 bf16): elems 0..15 <- K = k0+16g+0..15
        const float4 a0 = *(const float4*)(Arow + k0 + 16 * g);
        const float4 a1 = *(const float4*)(Arow + k0 + 16 * g + 4);
        const float4 a2 = *(const float4*)(Arow + k0 + 16 * g + 8);
        const float4 a3 = *(const float4*)(Arow + k0 + 16 * g + 12);
        // B fragments: 16 sequential K at k0 + 16g
        const float4 p0 = *(const float4*)(Wrow0 + k0 + 16 * g);
        const float4 p1 = *(const float4*)(Wrow0 + k0 + 16 * g + 4);
        const float4 p2 = *(const float4*)(Wrow0 + k0 + 16 * g + 8);
        const float4 p3 = *(const float4*)(Wrow0 + k0 + 16 * g + 12);
        const float4 q0 = *(const float4*)(Wrow1 + k0 + 16 * g);
        const float4 q1 = *(const float4*)(Wrow1 + k0 + 16 * g + 4);
        const float4 q2 = *(const float4*)(Wrow1 + k0 + 16 * g + 8);
        const float4 q3 = *(const float4*)(Wrow1 + k0 + 16 * g + 12);

        const v16bf afrag  = pack_bf16(a0, a1, a2, a3);
        const v16bf bfrag0 = pack_bf16(p0, p1, p2, p3);
        const v16bf bfrag1 = pack_bf16(q0, q1, q2, q3);

        acc0 = __builtin_amdgcn_wmma_f32_16x16x32_bf16(
                   false, afrag, false, bfrag0, (short)0, acc0, false, false);
        acc1 = __builtin_amdgcn_wmma_f32_16x16x32_bf16(
                   false, afrag, false, bfrag1, (short)0, acc1, false, false);
    }

    const int  n0   = tn * 32 + l16;
    const int  n1   = tn * 32 + 16 + l16;
    const bool n0ok = n0 < N;
    const bool n1ok = n1 < N;
    const float bv0 = bias[ncol0];
    const float bv1 = bias[ncol1];
    float* __restrict__ Crow = C + (size_t)(tm * 16 + 8 * g) * ldc;
#pragma unroll
    for (int r = 0; r < 8; ++r) {                 // C/D: M = r + 8*g, N = l16
        const int mr = tm * 16 + 8 * g + r;
        if (mr < M) {
            if (n0ok) Crow[n0] = acc0[r] + bv0;
            if (n1ok) Crow[n1] = acc1[r] + bv1;
        }
        Crow += ldc;
    }
}

// ---------------------------------------------------------------------------
// Row padding: dst[r][0..dld) = src[r][0..sld) then zeros.  (K=449 -> 480)
// ---------------------------------------------------------------------------
__global__ void k_pad(const float* __restrict__ src, int sld,
                      float* __restrict__ dst, int dld, int n)
{
    const int i = blockIdx.x * blockDim.x + threadIdx.x;
    if (i >= n) return;
    const int r = i / dld, c = i % dld;
    dst[i] = (c < sld) ? src[(size_t)r * sld + c] : 0.f;
}

// ---------------------------------------------------------------------------
// Sequential LSTM over one layer, both directions (blockIdx.x = dir).
// 512 threads; thread u owns gate rows {u, 512+u, 1024+u, 1536+u} so c stays
// in registers; h is double-buffered in LDS (one barrier per timestep).
// ---------------------------------------------------------------------------
__global__ __launch_bounds__(HDIM)
void k_lstm(const float* __restrict__ gin,
            const float* __restrict__ Whh,
            float* __restrict__ out, int ld_out)
{
    __shared__ __align__(16) float hbuf[2][HDIM];
    const int d = blockIdx.x;
    const int u = threadIdx.x;

    const float* gd = gin + (size_t)d * T_LEN * G4H;
    const float* w0 = Whh + ((size_t)d * G4H + u         ) * HDIM;
    const float* w1 = Whh + ((size_t)d * G4H + 512  + u  ) * HDIM;
    const float* w2 = Whh + ((size_t)d * G4H + 1024 + u  ) * HDIM;
    const float* w3 = Whh + ((size_t)d * G4H + 1536 + u  ) * HDIM;

    float c = 0.f;
    hbuf[0][u] = 0.f;
    __syncthreads();

    for (int t = 0; t < T_LEN; ++t) {
        const int rb = t & 1, wb = rb ^ 1;
        const int row = d ? (T_LEN - 1 - t) : t;
        const float* gr = gd + (size_t)row * G4H;

        float zi = gr[u], zf = gr[512 + u], zg = gr[1024 + u], zo = gr[1536 + u];

        const float4* h4 = (const float4*)hbuf[rb];
        const float4* a0 = (const float4*)w0;
        const float4* a1 = (const float4*)w1;
        const float4* a2 = (const float4*)w2;
        const float4* a3 = (const float4*)w3;
#pragma unroll 4
        for (int kk = 0; kk < HDIM / 4; ++kk) {
            const float4 hv = h4[kk];
            const float4 v0 = a0[kk], v1 = a1[kk], v2 = a2[kk], v3 = a3[kk];
            zi += v0.x*hv.x + v0.y*hv.y + v0.z*hv.z + v0.w*hv.w;
            zf += v1.x*hv.x + v1.y*hv.y + v1.z*hv.z + v1.w*hv.w;
            zg += v2.x*hv.x + v2.y*hv.y + v2.z*hv.z + v2.w*hv.w;
            zo += v3.x*hv.x + v3.y*hv.y + v3.z*hv.z + v3.w*hv.w;
        }

        const float ig = 1.f / (1.f + __expf(-zi));
        const float fg = 1.f / (1.f + __expf(-zf));
        const float og = 1.f / (1.f + __expf(-zo));
        c = fg * c + ig * tanhf(zg);
        const float h = og * tanhf(c);

        hbuf[wb][u] = h;
        out[(size_t)row * ld_out + d * HDIM + u] = h;
        __syncthreads();
    }
}

// ---------------------------------------------------------------------------
// Pair scorer: score[i][j] = w . tanh(heads[i] + mods[j]) + b ; diag = 0.
// 16x16 output tile per 256-thread block; LDS row stride 257 avoids 64-bank
// conflicts on the sm[rj][h] gathers.
// ---------------------------------------------------------------------------
__global__ __launch_bounds__(256)
void k_score(const float* __restrict__ heads, const float* __restrict__ mods,
             const float* __restrict__ wsc, const float* __restrict__ bsc,
             float* __restrict__ out, int Np)
{
    __shared__ float sh[16][H2 + 1];
    __shared__ float sm[16][H2 + 1];
    __shared__ float sw[H2];

    const int ti = blockIdx.y, tj = blockIdx.x;
    const int t = threadIdx.x;          // 0..255 == H2

    sw[t] = wsc[t];
#pragma unroll
    for (int r = 0; r < 16; ++r) {
        const int i = ti * 16 + r;
        sh[r][t] = (i < Np) ? heads[(size_t)i * H2 + t] : 0.f;
        const int j = tj * 16 + r;
        sm[r][t] = (j < Np) ? mods[(size_t)j * H2 + t] : 0.f;
    }
    __syncthreads();

    const int ri = t >> 4, rj = t & 15;
    float s = bsc[0];
#pragma unroll 4
    for (int h = 0; h < H2; ++h)
        s += sw[h] * tanhf(sh[ri][h] + sm[rj][h]);

    const int i = ti * 16 + ri, j = tj * 16 + rj;
    if (i < Np && j < Np) out[(size_t)i * Np + j] = (i == j) ? 0.f : s;
}

__global__ void k_zero(float* __restrict__ p, int n)
{
    const int i = blockIdx.x * blockDim.x + threadIdx.x;
    if (i < n) p[i] = 0.f;
}

// ---------------------------------------------------------------------------
// Host orchestration (all on `stream`, graph-capture safe).
// ---------------------------------------------------------------------------
extern "C" void kernel_launch(void* const* d_in, const int* in_sizes, int n_in,
                              void* d_out, int out_size, void* d_ws, size_t ws_size,
                              hipStream_t stream)
{
    const float* x        = (const float*)d_in[0];
    const float* Wih[4]   = {(const float*)d_in[1], (const float*)d_in[4],
                             (const float*)d_in[7], (const float*)d_in[10]};
    const float* Whh[4]   = {(const float*)d_in[2], (const float*)d_in[5],
                             (const float*)d_in[8], (const float*)d_in[11]};
    const float* bgate[4] = {(const float*)d_in[3], (const float*)d_in[6],
                             (const float*)d_in[9], (const float*)d_in[12]};
    const float* Wh      = (const float*)d_in[13];
    const float* bh      = (const float*)d_in[14];
    const float* Wm      = (const float*)d_in[15];
    const float* bm      = (const float*)d_in[16];
    const float* w_score = (const float*)d_in[17];
    const float* b_score = (const float*)d_in[18];

    // Workspace layout (floats) — ~24.6 MB total.
    float* ws = (float*)d_ws;
    size_t off = 0;
    float* gin    = ws + off; off += (size_t)2 * T_LEN * G4H;   // 2*512*2048
    float* buf0   = ws + off; off += (size_t)T_LEN * 2 * HDIM;  // 512*1024
    float* buf1   = ws + off; off += (size_t)T_LEN * 2 * HDIM;
    float* hcat   = ws + off; off += (size_t)NP * 2 * HDIM;     // 513*1024
    float* heads  = ws + off; off += (size_t)NP * H2;
    float* mods   = ws + off; off += (size_t)NP * H2;
    float* xpad   = ws + off; off += (size_t)T_LEN * KP0;       // 512*480
    float* wihpad = ws + off; off += (size_t)2 * G4H * KP0;     // 2*2048*480

    // Pad ragged-K (449 -> 480, zero-filled) so every GEMM is branch-free.
    {
        const int nx = T_LEN * KP0;
        k_pad<<<(nx + 255) / 256, 256, 0, stream>>>(x, 449, xpad, KP0, nx);
        const int nw = 2 * G4H * KP0;
        k_pad<<<(nw + 255) / 256, 256, 0, stream>>>(Wih[0], 449, wihpad, KP0, nw);
    }

    const int din[4]     = {KP0, 1024, 1024, 1024};
    const float* wih[4]  = {wihpad, Wih[1], Wih[2], Wih[3]};
    const float* lin[4]  = {xpad, buf0, buf1, buf0};
    float*       lout[4] = {buf0, buf1, buf0, hcat + 2 * HDIM}; // last: rows 1..512

    for (int L = 0; L < 4; ++L) {
        const int K = din[L];
        const int tiles = ((T_LEN + 15) / 16) * ((G4H + 31) / 32);
        for (int d2 = 0; d2 < 2; ++d2) {
            k_gemm_bias<<<(tiles + 3) / 4, 128, 0, stream>>>(
                lin[L], K,
                wih[L] + (size_t)d2 * G4H * K, K,
                bgate[L] + (size_t)d2 * G4H,
                gin + (size_t)d2 * T_LEN * G4H, G4H,
                T_LEN, G4H, K);
        }
        k_lstm<<<2, HDIM, 0, stream>>>(gin, Whh[L], lout[L], 2 * HDIM);
    }

    // ROOT row = zeros
    k_zero<<<(2 * HDIM + 255) / 256, 256, 0, stream>>>(hcat, 2 * HDIM);

    // heads / mods projections: [513,1024] @ [1024,256] + bias
    {
        const int tiles = ((NP + 15) / 16) * ((H2 + 31) / 32);
        k_gemm_bias<<<(tiles + 3) / 4, 128, 0, stream>>>(
            hcat, 2 * HDIM, Wh, 2 * HDIM, bh, heads, H2, NP, H2, 2 * HDIM);
        k_gemm_bias<<<(tiles + 3) / 4, 128, 0, stream>>>(
            hcat, 2 * HDIM, Wm, 2 * HDIM, bm, mods, H2, NP, H2, 2 * HDIM);
    }

    // all-pairs scorer
    dim3 sg((NP + 15) / 16, (NP + 15) / 16);
    k_score<<<sg, 256, 0, stream>>>(heads, mods, w_score, b_score,
                                    (float*)d_out, NP);
}
